// MultibandFrameAttention_17815524343846
// MI455X (gfx1250) — matrix operand order repaired and checked by reading, not compiled, not run on hardware
//
#include <hip/hip_runtime.h>
#include <hip/hip_bf16.h>

typedef __attribute__((ext_vector_type(16))) __bf16          v16bf;
typedef __attribute__((ext_vector_type(8)))  float           v8f;
typedef __attribute__((ext_vector_type(16))) unsigned short  v16us;
typedef __attribute__((ext_vector_type(8)))  unsigned short  v8us;

#define WMMA_BF16(A, B, C) \
  __builtin_amdgcn_wmma_f32_16x16x32_bf16(false, (A), false, (B), (short)0, (C), false, false)

__device__ __forceinline__ unsigned short f32_bf16(float f) {
  unsigned int u = __float_as_uint(f);
  return (unsigned short)((u + 0x7FFFu + ((u >> 16) & 1u)) >> 16);  // RNE
}

// A-fragment (16x32 bf16): lane = M row; element j = K 16h+j
// -> two contiguous 8-elem chunks.
__device__ __forceinline__ v16bf load_a_frag(const unsigned short* p, int h) {
  v8us a = *(const v8us*)(p + 16 * h);
  v8us b = *(const v8us*)(p + 16 * h + 8);
  v16us w = __builtin_shufflevector(a, b, 0,1,2,3,4,5,6,7,8,9,10,11,12,13,14,15);
  return __builtin_bit_cast(v16bf, w);
}
// B-fragment (32x16 bf16): lane = N col; element j = K j+16h -> 16 contiguous.
__device__ __forceinline__ v16bf load_b_frag(const unsigned short* p) {
  v16us w = *(const v16us*)p;
  return __builtin_bit_cast(v16bf, w);
}

// ---------------------------------------------------------------------------
// Kernel 1: fused q/k/v 3x3 conv as tap-pair GEMM.
//   out[co,pix] = sum_{s=0..4} Wpair_s(16co x 32k) * Xpair_s(32k x 16pix),
//   k = half*16 + ci, tap t = 2s+half (t==9 -> zeros).
// LDS x-tile is channel-minor: xtile[(dy*130+xx)*16 + ci]; row dy==3 zeroed,
// so the B-fragment is ONE contiguous 32B LDS load per lane.
// Writes q,k as [inst][frame(512)][d(32)] bf16 and v transposed [inst][d][frame].
// inst = ((b*16+c)*8 + head), head = y/32, d = y%32.
// ---------------------------------------------------------------------------
__global__ __launch_bounds__(256) void qkv_conv_kernel(
    const float* __restrict__ x,
    const float* __restrict__ wq, const float* __restrict__ bq,
    const float* __restrict__ wk, const float* __restrict__ bk,
    const float* __restrict__ wv, const float* __restrict__ bv,
    unsigned short* __restrict__ q_ws, unsigned short* __restrict__ k_ws,
    unsigned short* __restrict__ vT_ws) {
  __shared__ __align__(32) unsigned short xtile[520 * 16];      // [dy*130+xx][ci]
  __shared__ __align__(32) unsigned short wlds[3 * 16 * 160];   // [set][co][s*32+half*16+ci]

  const int tid = threadIdx.x;
  const int xt = blockIdx.x, y = blockIdx.y, b = blockIdx.z;
  const int x0 = xt * 128;

  // Stage x tile (coalesced in gx), transposed write (channel-minor).
  for (int i = tid; i < 16 * 390; i += 256) {
    int ci = i / 390, rem = i % 390, dy = rem / 130, xx = rem % 130;
    int gy = y + dy - 1, gx = x0 + xx - 1;
    float v = 0.f;
    if (gy >= 0 && gy < 256 && gx >= 0 && gx < 512)
      v = x[((b * 16 + ci) * 256 + gy) * 512 + gx];
    xtile[(dy * 130 + xx) * 16 + ci] = f32_bf16(v);
  }
  for (int i = tid; i < 130 * 16; i += 256) xtile[390 * 16 + i] = 0;  // tap-9 zeros

  // Weights: kk = s*32 + half*16 + ci, tap t = 2s+half, value w[co][ci][t].
  for (int i = tid; i < 3 * 16 * 160; i += 256) {
    int set = i / 2560, rr = i % 2560, co = rr / 160, kk = rr % 160;
    int s = kk / 32, r2 = kk % 32, half = r2 / 16, ci = r2 % 16, t = 2 * s + half;
    const float* wsrc = (set == 0) ? wq : (set == 1) ? wk : wv;
    wlds[i] = f32_bf16(t < 9 ? wsrc[(co * 16 + ci) * 9 + t] : 0.f);
  }
  __syncthreads();

  const int wave = tid >> 5, lane = tid & 31, h = lane >> 4, qp = lane & 15;
  const int xl = wave * 16 + qp;  // pixel column within 128-wide tile

  v8f cq = {}, ck = {}, cv = {};
#pragma unroll
  for (int s = 0; s < 5; s++) {
    int tap = 2 * s + h;                      // tap 9 -> dy=3 (zero row)
    int dy = tap / 3, dx = tap % 3;
    v16bf B = load_b_frag(&xtile[(dy * 130 + xl + dx) * 16]);
    cq = WMMA_BF16(load_a_frag(&wlds[0 * 2560 + qp * 160 + s * 32], h), B, cq);
    ck = WMMA_BF16(load_a_frag(&wlds[1 * 2560 + qp * 160 + s * 32], h), B, ck);
    cv = WMMA_BF16(load_a_frag(&wlds[2 * 2560 + qp * 160 + s * 32], h), B, cv);
  }

  const int head = y >> 5, dd = y & 31, frame = x0 + xl;
#pragma unroll
  for (int r = 0; r < 8; r++) {
    int co = r + 8 * h;
    int inst = (b * 16 + co) * 8 + head;
    q_ws[(inst * 512 + frame) * 32 + dd] = f32_bf16(cq[r] + bq[co]);
    k_ws[(inst * 512 + frame) * 32 + dd] = f32_bf16(ck[r] + bk[co]);
    vT_ws[(inst * 32 + dd) * 512 + frame] = f32_bf16(cv[r] + bv[co]);
  }
}

// ---------------------------------------------------------------------------
// Kernel 2: flash attention per instance (512 inst, seq 512, d 32), transposed:
// S^T = K Q^T  (per-query softmax stats are lane-local), O^T = V^T P^T.
// ---------------------------------------------------------------------------
__global__ __launch_bounds__(256) void attn_kernel(
    const unsigned short* __restrict__ q_ws,
    const unsigned short* __restrict__ k_ws,
    const unsigned short* __restrict__ vT_ws,
    unsigned short* __restrict__ a_ws) {
  const int tid = threadIdx.x;
  const int inst = blockIdx.y;
  const int wave = tid >> 5, lane = tid & 31, h = lane >> 4, qc = lane & 15;
  const int qt = blockIdx.x * 8 + wave;  // 16-query tile index (0..31)

  const unsigned short* qb = q_ws + (size_t)inst * 16384;
  const unsigned short* kb = k_ws + (size_t)inst * 16384;
  const unsigned short* vb = vT_ws + (size_t)inst * 16384;

  // B = Q^T fragment (loop-invariant): lane = query col, elements = d j+16h
  v16bf BQ = load_b_frag(qb + (qt * 16 + qc) * 32 + 16 * h);

  float m = -1e30f, lsum = 0.f;
  v8f o0 = {}, o1 = {};
  const float SC = 0.0625f;  // 1/sqrt(h=256)

  for (int u0 = 0; u0 < 512; u0 += 32) {
    if (u0 + 32 < 512) {
      __builtin_prefetch(kb + (u0 + 32 + qc) * 32, 0, 1);
      __builtin_prefetch(vb + qc * 512 + u0 + 32, 0, 1);
    }
    // S^T tiles: keys u0..u0+15 and u0+16..u0+31 vs 16 queries
    v16bf A0 = load_a_frag(kb + (u0 + qc) * 32, h);
    v16bf A1 = load_a_frag(kb + (u0 + 16 + qc) * 32, h);
    v8f z = {};
    v8f t0 = WMMA_BF16(A0, BQ, z);
    v8f t1 = WMMA_BF16(A1, BQ, z);

    float mt = -1e30f;
#pragma unroll
    for (int r = 0; r < 8; r++) {
      t0[r] *= SC; t1[r] *= SC;
      mt = fmaxf(mt, fmaxf(t0[r], t1[r]));
    }
    mt = fmaxf(mt, __shfl_xor(mt, 16, 32));  // combine halves -> 32-key max
    float mn = fmaxf(m, mt);
    float alpha = __expf(m - mn);
    m = mn;

    float p0[8], p1[8], sum = 0.f;
#pragma unroll
    for (int r = 0; r < 8; r++) {
      p0[r] = __expf(t0[r] - mn);
      p1[r] = __expf(t1[r] - mn);
      sum += p0[r] + p1[r];
    }
    sum += __shfl_xor(sum, 16, 32);
    lsum = lsum * alpha + sum;
#pragma unroll
    for (int r = 0; r < 8; r++) { o0[r] *= alpha; o1[r] *= alpha; }

    // P^T as B-fragment: lane-half h needs keys 16h..16h+15 = own/partner mix
    v16us pe;
#pragma unroll
    for (int j = 0; j < 8; j++) {
      float xp0 = __shfl_xor(p0[j], 16, 32);
      float xp1 = __shfl_xor(p1[j], 16, 32);
      pe[j]     = f32_bf16(h ? xp1 : p0[j]);
      pe[j + 8] = f32_bf16(h ? p1[j] : xp0);
    }
    v16bf BPT = __builtin_bit_cast(v16bf, pe);

    // V^T A-fragments (M = d rows, K = keys of this step)
    v16bf VA0 = load_a_frag(vb + qc * 512 + u0, h);
    v16bf VA1 = load_a_frag(vb + (16 + qc) * 512 + u0, h);
    o0 = WMMA_BF16(VA0, BPT, o0);
    o1 = WMMA_BF16(VA1, BPT, o1);
  }

  float inv = 1.f / lsum;
  const int bc = inst >> 3, head = inst & 7;
  const int frame = qt * 16 + qc;
#pragma unroll
  for (int r = 0; r < 8; r++) {
    int d0 = r + 8 * h, d1 = 16 + r + 8 * h;
    a_ws[((size_t)bc * 256 + head * 32 + d0) * 512 + frame] = f32_bf16(o0[r] * inv);
    a_ws[((size_t)bc * 256 + head * 32 + d1) * 512 + frame] = f32_bf16(o1[r] * inv);
  }
}

// ---------------------------------------------------------------------------
// Kernel 3: output 3x3 conv, same tap-pair GEMM (bf16 in from a_ws, f32 out).
// ---------------------------------------------------------------------------
__global__ __launch_bounds__(256) void out_conv_kernel(
    const unsigned short* __restrict__ a_ws,
    const float* __restrict__ wo, const float* __restrict__ bo,
    float* __restrict__ out) {
  __shared__ __align__(32) unsigned short xtile[520 * 16];
  __shared__ __align__(32) unsigned short wlds[16 * 160];

  const int tid = threadIdx.x;
  const int xt = blockIdx.x, y = blockIdx.y, b = blockIdx.z;
  const int x0 = xt * 128;

  for (int i = tid; i < 16 * 390; i += 256) {
    int ci = i / 390, rem = i % 390, dy = rem / 130, xx = rem % 130;
    int gy = y + dy - 1, gx = x0 + xx - 1;
    unsigned short v = 0;
    if (gy >= 0 && gy < 256 && gx >= 0 && gx < 512)
      v = a_ws[((size_t)(b * 16 + ci) * 256 + gy) * 512 + gx];
    xtile[(dy * 130 + xx) * 16 + ci] = v;
  }
  for (int i = tid; i < 130 * 16; i += 256) xtile[390 * 16 + i] = 0;

  for (int i = tid; i < 16 * 160; i += 256) {
    int co = i / 160, kk = i % 160;
    int s = kk / 32, r2 = kk % 32, half = r2 / 16, ci = r2 % 16, t = 2 * s + half;
    wlds[i] = f32_bf16(t < 9 ? wo[(co * 16 + ci) * 9 + t] : 0.f);
  }
  __syncthreads();

  const int wave = tid >> 5, lane = tid & 31, h = lane >> 4, qp = lane & 15;
  const int xl = wave * 16 + qp;
  v8f acc = {};
#pragma unroll
  for (int s = 0; s < 5; s++) {
    int tap = 2 * s + h;
    int dy = tap / 3, dx = tap % 3;
    v16bf B = load_b_frag(&xtile[(dy * 130 + xl + dx) * 16]);
    acc = WMMA_BF16(load_a_frag(&wlds[qp * 160 + s * 32], h), B, acc);
  }
#pragma unroll
  for (int r = 0; r < 8; r++) {
    int co = r + 8 * h;
    out[(((size_t)b * 16 + co) * 256 + y) * 512 + x0 + xl] = acc[r] + bo[co];
  }
}

// ---------------------------------------------------------------------------
extern "C" void kernel_launch(void* const* d_in, const int* in_sizes, int n_in,
                              void* d_out, int out_size, void* d_ws, size_t ws_size,
                              hipStream_t stream) {
  const float* x  = (const float*)d_in[0];
  const float* wq = (const float*)d_in[1];
  const float* bq = (const float*)d_in[2];
  const float* wk = (const float*)d_in[3];
  const float* bk = (const float*)d_in[4];
  const float* wv = (const float*)d_in[5];
  const float* bv = (const float*)d_in[6];
  const float* wo = (const float*)d_in[7];
  const float* bo = (const float*)d_in[8];
  float* out = (float*)d_out;

  unsigned short* ws = (unsigned short*)d_ws;
  const size_t N = (size_t)512 * 16384;  // 8,388,608 bf16 per tensor
  unsigned short* q_ws  = ws;
  unsigned short* k_ws  = ws + N;
  unsigned short* vT_ws = ws + 2 * N;
  unsigned short* a_ws  = ws + 3 * N;

  qkv_conv_kernel<<<dim3(4, 256, 4), 256, 0, stream>>>(
      x, wq, bq, wk, bk, wv, bv, q_ws, k_ws, vT_ws);
  attn_kernel<<<dim3(4, 512), 256, 0, stream>>>(q_ws, k_ws, vT_ws, a_ws);
  out_conv_kernel<<<dim3(4, 256, 4), 256, 0, stream>>>(a_ws, wo, bo, out);
}
